// hasd_77635828842940
// MI455X (gfx1250) — matrix operand order repaired and checked by reading, not compiled, not run on hardware
//
#include <hip/hip_runtime.h>
#include <cstdint>
#include <cstddef>

// ---------------------------------------------------------------------------
// Problem constants (from the reference): B=64, C=128, L=1024, scales 8/4/2/1
// ---------------------------------------------------------------------------
constexpr int kB = 64;
constexpr int kC = 128;
constexpr int kL = 1024;
constexpr int kCL = kC * kL;                  // 131072 features for batchnorm
constexpr size_t kMN = (size_t)kB * kC * kL;  // 8,388,608 elements (M=8192 x N=1024)
constexpr int kM = kB * kC;                   // 8192 GEMM rows

typedef __bf16 bf16_t;
typedef __attribute__((ext_vector_type(16))) __bf16 v16bf;
typedef __attribute__((ext_vector_type(8)))  __bf16 v8bf;
typedef __attribute__((ext_vector_type(8)))  float  v8f;

__device__ __forceinline__ float sigmoidf_(float x) { return 1.0f / (1.0f + __expf(-x)); }
__device__ __forceinline__ float geluf_(float x) {
  // exact gelu: 0.5*x*(1+erf(x/sqrt(2)))  (reference uses approximate=False)
  return 0.5f * x * (1.0f + erff(x * 0.70710678118654752440f));
}

// CDNA5 async global->LDS copy (ASYNCcnt-tracked, no VGPR round trip).
// lds_off: wave-relative LDS byte offset (low 32 bits of the generic pointer).
__device__ __forceinline__ void async_lds_b128(unsigned lds_off, const void* gaddr) {
  asm volatile("global_load_async_to_lds_b128 %0, %1, off"
               :: "v"(lds_off), "v"(gaddr)
               : "memory");
}
__device__ __forceinline__ void wait_asynccnt0() {
  asm volatile("s_wait_asynccnt 0x0" ::: "memory");
}

// ---------------------------------------------------------------------------
// 1) BatchNorm statistics over batch axis (one thread per feature f in [0,CL))
// ---------------------------------------------------------------------------
__global__ void bn_stats_kernel(const float* __restrict__ x,
                                float* __restrict__ mu, float* __restrict__ rinv) {
  int f = blockIdx.x * blockDim.x + threadIdx.x;
  if (f >= kCL) return;
  float s1 = 0.f, s2 = 0.f;
  for (int b = 0; b < kB; ++b) {
    float v = x[(size_t)b * kCL + f];
    s1 += v; s2 += v * v;
  }
  float m = s1 * (1.0f / kB);
  float var = fmaxf(s2 * (1.0f / kB) - m * m, 0.0f);
  mu[f] = m;
  rinv[f] = rsqrtf(var + 1e-5f);
}

__global__ void bn_apply_kernel(const float* __restrict__ x,
                                const float* __restrict__ mu, const float* __restrict__ rinv,
                                const float* __restrict__ gg, const float* __restrict__ bb,
                                float* __restrict__ xn) {
  size_t e = (size_t)blockIdx.x * blockDim.x + threadIdx.x;
  if (e >= kMN) return;
  int f = (int)(e % kCL);
  xn[e] = (x[e] - mu[f]) * rinv[f] * gg[f] + bb[f];
}

// ---------------------------------------------------------------------------
// 2) Depthwise strided conv (downsample). One thread per output element.
// ---------------------------------------------------------------------------
__global__ void dwconv_kernel(const float* __restrict__ xn, const float* __restrict__ cw,
                              float* __restrict__ df, bf16_t* __restrict__ db,
                              int s, int Ls) {
  size_t idx = (size_t)blockIdx.x * blockDim.x + threadIdx.x;
  size_t total = (size_t)kM * Ls;
  if (idx >= total) return;
  int t = (int)(idx % Ls);
  size_t bc = idx / Ls;            // b*C + c
  int c = (int)(bc % kC);
  const float* xr = xn + bc * (size_t)kL;
  const float* wr = cw + (size_t)c * (2 * s);
  int p0 = t * s - (s >> 1);
  float acc = 0.f;
  for (int k = 0; k < 2 * s; ++k) {
    int p = p0 + k;
    if (p >= 0 && p < kL) acc += xr[p] * wr[k];
  }
  df[idx] = acc;
  db[idx] = (bf16_t)acc;
}

// ---------------------------------------------------------------------------
// 3) EMA: trend[t] = cumsum(x*w_num)[t] / max(cumsum(base)[t], 1e-12)
//    Two plain cumsums -> wave32 inclusive scans (one wave per (b,c) row).
// ---------------------------------------------------------------------------
template <bool WF32, bool WBF16>
__global__ void ema_kernel(const float* __restrict__ x, const float* __restrict__ al,
                           float* __restrict__ outF, bf16_t* __restrict__ outB, int Ls) {
  int lane = threadIdx.x & 31;
  int row = blockIdx.x * (blockDim.x >> 5) + (threadIdx.x >> 5);
  if (row >= kM) return;
  int c = row % kC;
  float alpha = sigmoidf_(al[c]);
  float r = 1.0f - alpha;
  float lr = __logf(r);
  const float* xr = x + (size_t)row * Ls;
  float cn = 0.0f, cd = 0.0f;                 // carried cumsum of num / den
  for (int t0 = 0; t0 < Ls; t0 += 32) {
    int t = t0 + lane;
    float p = (float)(Ls - 1 - t);
    float base = __expf(lr * p);              // underflows to 0 like the reference
    float w = base * ((t == 0) ? 1.0f : alpha);
    float y = xr[t] * w;
    float d = base;
    #pragma unroll
    for (int sd = 1; sd < 32; sd <<= 1) {     // wave32 inclusive scan
      float yy = __shfl_up(y, sd, 32);
      float dd = __shfl_up(d, sd, 32);
      if (lane >= sd) { y += yy; d += dd; }
    }
    float num = cn + y;
    float den = cd + d;
    float tr = num / fmaxf(den, 1e-12f);
    if (WF32)  outF[(size_t)row * Ls + t] = tr;
    if (WBF16) outB[(size_t)row * Ls + t] = (bf16_t)tr;
    cn += __shfl(y, 31, 32);
    cd += __shfl(d, 31, 32);
  }
}

// ---------------------------------------------------------------------------
// 4) bf16 WMMA GEMM:  out[m,n] = act( sum_k A[m,k]*W[n,k] + bias[n] )
//    A: M x K row-major bf16 (K contiguous); W: N x K row-major bf16.
//    128x128 block tile, 256 threads (8 waves), each wave a 32x64 tile =
//    2x4 v_wmma_f32_16x16x32_bf16 accumulators. K stepped 32 at a time.
//    Double-buffered LDS fed by global_load_async_to_lds_b128. K-loop is
//    unrolled by 2 (numK = K/32 is always even here) so the ping-pong buffer
//    is selected statically -- no cndmask address selection, no buffer-base
//    multiplies. One s_wait_asynccnt+barrier per K-step: barrier N proves all
//    waves finished their ds reads of the buffer that step N+1 overwrites
//    (the consuming WMMA forced s_wait_dscnt before the barrier signal).
// ---------------------------------------------------------------------------
struct Frag { union { v16bf v; v8bf h[2]; }; };

template <bool GELU, bool OUT_BF16>
__global__ __launch_bounds__(256)
void gemm_wmma(const bf16_t* __restrict__ A, const bf16_t* __restrict__ W,
               const float* __restrict__ bias, float* __restrict__ outF,
               bf16_t* __restrict__ outB, int M, int N, int K) {
  // +8 halves padding: row stride 80B = 20 dwords -> conflict-free 16-row
  // fragment reads, still 16-byte aligned for ds_load_b128.
  __shared__ bf16_t As[2][128][40];
  __shared__ bf16_t Bs[2][128][40];

  const int tid = threadIdx.x;
  const int lane = tid & 31;
  const int wid = tid >> 5;
  const int bm = blockIdx.y * 128;
  const int bn = blockIdx.x * 128;
  const int wm = (wid & 3) * 32;   // 4 waves along M
  const int wn = (wid >> 2) * 64;  // 2 waves along N
  const int rsel = lane & 15;
  const int abase = (lane < 16) ? 0 : 8;   // A 16-bit layout (ISA 7.12.2)
  const int kbB = (lane < 16) ? 0 : 8;     // B 16-bit layout: same K order as A

  // Loader mapping: each thread owns 4x16B chunks per K-tile (2 A, 2 W):
  //   rows r0 and r0+64, 16B column c0. LDS dests are fixed per buffer;
  //   global pointers advance by 64 elements (128B) per unrolled iteration.
  const int r0 = tid >> 2;            // 0..63
  const int c0 = (tid & 3) << 3;      // 0,8,16,24 (bf16 elements)
  const bf16_t* gA0 = A + (size_t)(bm + r0) * K + c0;
  const bf16_t* gA1 = A + (size_t)(bm + r0 + 64) * K + c0;
  const bf16_t* gW0 = W + (size_t)(bn + r0) * K + c0;
  const bf16_t* gW1 = W + (size_t)(bn + r0 + 64) * K + c0;
  unsigned lA0[2], lA1[2], lW0[2], lW1[2];
  #pragma unroll
  for (int b = 0; b < 2; ++b) {
    lA0[b] = (unsigned)(uintptr_t)&As[b][r0][c0];
    lA1[b] = (unsigned)(uintptr_t)&As[b][r0 + 64][c0];
    lW0[b] = (unsigned)(uintptr_t)&Bs[b][r0][c0];
    lW1[b] = (unsigned)(uintptr_t)&Bs[b][r0 + 64][c0];
  }

  v8f acc[2][4];
  #pragma unroll
  for (int i = 0; i < 2; ++i)
    #pragma unroll
    for (int j = 0; j < 4; ++j)
      #pragma unroll
      for (int e = 0; e < 8; ++e) acc[i][j][e] = 0.0f;

  // issue one K-tile's async copies into buffer `b` from offset `koff`
  auto issue = [&](int b, int koff) {
    async_lds_b128(lA0[b], gA0 + koff);
    async_lds_b128(lA1[b], gA1 + koff);
    async_lds_b128(lW0[b], gW0 + koff);
    async_lds_b128(lW1[b], gW1 + koff);
  };

  // 12 ds_load_b128 fragment loads + 8 WMMAs for one 32-wide K-step
  auto compute_tile = [&](const bf16_t (*At)[40], const bf16_t (*Wt)[40]) {
    Frag afr[2], bfr[4];
    #pragma unroll
    for (int sm = 0; sm < 2; ++sm) {
      int row = wm + sm * 16 + rsel;
      afr[sm].h[0] = *(const v8bf*)&At[row][abase];       // K = base..base+7
      afr[sm].h[1] = *(const v8bf*)&At[row][16 + abase];  // K = 16+base..
    }
    #pragma unroll
    for (int sn = 0; sn < 4; ++sn) {
      int nrow = wn + sn * 16 + rsel;
      bfr[sn].h[0] = *(const v8bf*)&Wt[nrow][kbB];        // K = kb..kb+7
      bfr[sn].h[1] = *(const v8bf*)&Wt[nrow][kbB + 16];   // K = 16+kb..16+kb+7
    }
    #pragma unroll
    for (int sm = 0; sm < 2; ++sm)
      #pragma unroll
      for (int sn = 0; sn < 4; ++sn)
        acc[sm][sn] = __builtin_amdgcn_wmma_f32_16x16x32_bf16(
            false, afr[sm].v, false, bfr[sn].v, (short)0, acc[sm][sn], false, false);
  };

  const int numK = K >> 5;  // K in {128,256,512,1024} -> numK even

  // prologue: stage tile 0 into buffer 0
  issue(0, 0);
  wait_asynccnt0();
  __syncthreads();

  for (int kt = 0; kt < numK; kt += 2) {
    // tile kt in buf0; prefetch kt+1 into buf1 (kt+1 < numK since numK even)
    issue(1, 32);
    compute_tile(As[0], Bs[0]);
    wait_asynccnt0();
    __syncthreads();

    // tile kt+1 in buf1; prefetch kt+2 into buf0
    if (kt + 2 < numK) issue(0, 64);
    compute_tile(As[1], Bs[1]);
    wait_asynccnt0();
    __syncthreads();

    gA0 += 64; gA1 += 64; gW0 += 64; gW1 += 64;
  }

  // ---- epilogue: D layout -> row = v + (lane>=16 ? 8 : 0), col = lane&15 ----
  const int rowOff = (lane >> 4) << 3;
  #pragma unroll
  for (int sm = 0; sm < 2; ++sm)
    #pragma unroll
    for (int sn = 0; sn < 4; ++sn) {
      int col = bn + wn + sn * 16 + rsel;
      float bv = bias[col];
      #pragma unroll
      for (int v = 0; v < 8; ++v) {
        int row = bm + wm + sm * 16 + rowOff + v;
        float val = acc[sm][sn][v] + bv;
        if (GELU) val = geluf_(val);
        size_t o = (size_t)row * N + col;
        if (OUT_BF16) outB[o] = (bf16_t)val;
        else          outF[o] = val;
      }
    }
}

// ---------------------------------------------------------------------------
// 5) Gate path: per-(b,c) mean of season, per-b dot with gw, sigmoid, normalize
// ---------------------------------------------------------------------------
__global__ void mean_season_kernel(const float* __restrict__ up, const float* __restrict__ tup,
                                   float* __restrict__ ms) {
  int bc = blockIdx.x * blockDim.x + threadIdx.x;
  if (bc >= kM) return;
  const float* u = up + (size_t)bc * kL;
  const float* t = tup + (size_t)bc * kL;
  float s = 0.f;
  for (int l = 0; l < kL; ++l) s += u[l] - t[l];
  ms[bc] = s * (1.0f / kL);
}

__global__ void gate_kernel(const float* __restrict__ ms,
                            const float* gw8, const float* gb8,
                            const float* gw4, const float* gb4,
                            const float* gw2, const float* gb2,
                            const float* gw1, const float* gb1,
                            float* __restrict__ g) {
  __shared__ float red[kC];
  __shared__ float gsh[4];
  int b = blockIdx.x, c = threadIdx.x;
  const float* gws[4] = {gw8, gw4, gw2, gw1};
  const float* gbs[4] = {gb8, gb4, gb2, gb1};
  for (int sc = 0; sc < 4; ++sc) {
    red[c] = ms[(size_t)sc * kM + b * kC + c] * gws[sc][c];
    __syncthreads();
    for (int st = kC / 2; st > 0; st >>= 1) {
      if (c < st) red[c] += red[c + st];
      __syncthreads();
    }
    if (c == 0) gsh[sc] = sigmoidf_(red[0] + gbs[sc][0]);
    __syncthreads();
  }
  if (c == 0) {
    float tot = gsh[0] + gsh[1] + gsh[2] + gsh[3] + 1e-6f;
    for (int sc = 0; sc < 4; ++sc) g[b * 4 + sc] = gsh[sc] / tot;
  }
}

// ---------------------------------------------------------------------------
// 6) Weighted combine: season_out then trend_out (tuple return order)
// ---------------------------------------------------------------------------
__global__ void combine_kernel(const float* __restrict__ up8, const float* __restrict__ tup8,
                               const float* __restrict__ up4, const float* __restrict__ tup4,
                               const float* __restrict__ up2, const float* __restrict__ tup2,
                               const float* __restrict__ xn, const float* __restrict__ trend1,
                               const float* __restrict__ g, float* __restrict__ out) {
  size_t e = (size_t)blockIdx.x * blockDim.x + threadIdx.x;
  if (e >= kMN) return;
  int b = (int)(e / (size_t)kCL);
  float g0 = g[b * 4 + 0], g1 = g[b * 4 + 1], g2 = g[b * 4 + 2], g3 = g[b * 4 + 3];
  float t8 = tup8[e], t4 = tup4[e], t2 = tup2[e], t1 = trend1[e];
  float s8 = up8[e] - t8, s4 = up4[e] - t4, s2 = up2[e] - t2, s1 = xn[e] - t1;
  out[e]       = g0 * s8 + g1 * s4 + g2 * s2 + g3 * s1;
  out[kMN + e] = g0 * t8 + g1 * t4 + g2 * t2 + g3 * t1;
}

// ---------------------------------------------------------------------------
// misc small kernels
// ---------------------------------------------------------------------------
__global__ void cvt_bf16_kernel(const float* __restrict__ in, bf16_t* __restrict__ out, size_t n) {
  size_t e = (size_t)blockIdx.x * blockDim.x + threadIdx.x;
  if (e < n) out[e] = (bf16_t)in[e];
}

// ---------------------------------------------------------------------------
// Host launcher
// ---------------------------------------------------------------------------
static inline size_t align256(size_t x) { return (x + 255) & ~(size_t)255; }

extern "C" void kernel_launch(void* const* d_in, const int* in_sizes, int n_in,
                              void* d_out, int out_size, void* d_ws, size_t ws_size,
                              hipStream_t stream) {
  (void)in_sizes; (void)n_in; (void)out_size; (void)ws_size;
  const float* x    = (const float*)d_in[0];
  const float* bn_g = (const float*)d_in[1];
  const float* bn_b = (const float*)d_in[2];
  const float* cw[3] = {(const float*)d_in[3],  (const float*)d_in[8],  (const float*)d_in[13]};
  const float* w1[3] = {(const float*)d_in[4],  (const float*)d_in[9],  (const float*)d_in[14]};
  const float* b1[3] = {(const float*)d_in[5],  (const float*)d_in[10], (const float*)d_in[15]};
  const float* w2[3] = {(const float*)d_in[6],  (const float*)d_in[11], (const float*)d_in[16]};
  const float* b2[3] = {(const float*)d_in[7],  (const float*)d_in[12], (const float*)d_in[17]};
  const float* gw[4] = {(const float*)d_in[18], (const float*)d_in[21], (const float*)d_in[24], (const float*)d_in[27]};
  const float* gb[4] = {(const float*)d_in[19], (const float*)d_in[22], (const float*)d_in[25], (const float*)d_in[28]};
  const float* al[4] = {(const float*)d_in[20], (const float*)d_in[23], (const float*)d_in[26], (const float*)d_in[29]};

  char* ws = (char*)d_ws;
  size_t off = 0;
  auto alloc = [&](size_t bytes) -> void* {
    void* p = ws + off;
    off = align256(off + bytes);
    return p;
  };
  float* mu     = (float*)alloc((size_t)kCL * 4);
  float* rinv   = (float*)alloc((size_t)kCL * 4);
  float* xn     = (float*)alloc(kMN * 4);
  float* trend1 = (float*)alloc(kMN * 4);
  float* up[3]; float* tup[3];
  for (int i = 0; i < 3; ++i) { up[i] = (float*)alloc(kMN * 4); tup[i] = (float*)alloc(kMN * 4); }
  bf16_t* hbuf  = (bf16_t*)alloc(kMN * 2);
  bf16_t* w1bf  = (bf16_t*)alloc((size_t)kL * 512 * 2);   // max Ls = 512
  bf16_t* w2bf  = (bf16_t*)alloc((size_t)kL * kL * 2);
  bf16_t* dbf   = (bf16_t*)alloc((kMN / 2) * 2);          // max down size (s=2)
  bf16_t* tdsbf = (bf16_t*)alloc((kMN / 2) * 2);
  float*  downf = (float*)alloc((kMN / 2) * 4);
  float*  ms    = (float*)alloc((size_t)4 * kM * 4);
  float*  gvec  = (float*)alloc((size_t)kB * 4 * 4);

  // 1) batchnorm
  bn_stats_kernel<<<kCL / 256, 256, 0, stream>>>(x, mu, rinv);
  bn_apply_kernel<<<(unsigned)((kMN + 255) / 256), 256, 0, stream>>>(x, mu, rinv, bn_g, bn_b, xn);

  // 2) scale s=1: trend = EMA(xn), up = xn (identity)
  ema_kernel<true, false><<<kM / 8, 256, 0, stream>>>(xn, al[3], trend1, nullptr, kL);

  // 3) scales 8,4,2
  const int svals[3] = {8, 4, 2};
  dim3 ggrid(kL / 128, kM / 128);  // (8, 64)
  for (int i = 0; i < 3; ++i) {
    int s = svals[i];
    int Ls = kL / s;
    size_t w1n = (size_t)kL * Ls;
    cvt_bf16_kernel<<<(unsigned)((w1n + 255) / 256), 256, 0, stream>>>(w1[i], w1bf, w1n);
    cvt_bf16_kernel<<<(unsigned)(((size_t)kL * kL + 255) / 256), 256, 0, stream>>>(w2[i], w2bf, (size_t)kL * kL);

    size_t ndown = (size_t)kM * Ls;
    dwconv_kernel<<<(unsigned)((ndown + 255) / 256), 256, 0, stream>>>(xn, cw[i], downf, dbf, s, Ls);
    ema_kernel<false, true><<<kM / 8, 256, 0, stream>>>(downf, al[i], nullptr, tdsbf, Ls);

    // up path:  h = gelu(down @ w1^T + b1) ; up = h @ w2^T + b2
    gemm_wmma<true,  true ><<<ggrid, 256, 0, stream>>>(dbf,   w1bf, b1[i], nullptr, hbuf, kM, kL, Ls);
    gemm_wmma<false, false><<<ggrid, 256, 0, stream>>>(hbuf,  w2bf, b2[i], up[i],  nullptr, kM, kL, kL);
    // trend path
    gemm_wmma<true,  true ><<<ggrid, 256, 0, stream>>>(tdsbf, w1bf, b1[i], nullptr, hbuf, kM, kL, Ls);
    gemm_wmma<false, false><<<ggrid, 256, 0, stream>>>(hbuf,  w2bf, b2[i], tup[i], nullptr, kM, kL, kL);
  }

  // 4) gates (scale order 8,4,2,1)
  mean_season_kernel<<<kM / 256, 256, 0, stream>>>(up[0], tup[0], ms + 0 * (size_t)kM);
  mean_season_kernel<<<kM / 256, 256, 0, stream>>>(up[1], tup[1], ms + 1 * (size_t)kM);
  mean_season_kernel<<<kM / 256, 256, 0, stream>>>(up[2], tup[2], ms + 2 * (size_t)kM);
  mean_season_kernel<<<kM / 256, 256, 0, stream>>>(xn,    trend1, ms + 3 * (size_t)kM);
  gate_kernel<<<kB, kC, 0, stream>>>(ms, gw[0], gb[0], gw[1], gb[1], gw[2], gb[2], gw[3], gb[3], gvec);

  // 5) combine -> d_out = [season_out | trend_out]
  combine_kernel<<<(unsigned)((kMN + 255) / 256), 256, 0, stream>>>(
      up[0], tup[0], up[1], tup[1], up[2], tup[2], xn, trend1, gvec, (float*)d_out);
}
